// Attention_28896539968379
// MI455X (gfx1250) — compile-verified, matrix operand order checked by reading, not run on hardware
//
#include <hip/hip_runtime.h>
#include <hip/hip_bf16.h>
#include <math.h>

// ---------------------------------------------------------------------------
// CDNA5 (gfx1250) wave32 WMMA attention pipeline, async-LDS double-buffered.
// ---------------------------------------------------------------------------

typedef __attribute__((ext_vector_type(16))) __bf16 v16bf;
typedef __attribute__((ext_vector_type(8)))  float  v8f;

#ifndef __has_builtin
#define __has_builtin(x) 0
#endif

#if defined(__AMDGCN__) && __has_builtin(__builtin_amdgcn_global_load_async_to_lds_b128)
#define HAVE_ASYNC_LDS 1
#else
#define HAVE_ASYNC_LDS 0
#endif

#if HAVE_ASYNC_LDS
typedef __attribute__((vector_size(16))) int v4i;
typedef __attribute__((address_space(1))) v4i* v4i_gptr;  // global int4*
typedef __attribute__((address_space(3))) v4i* v4i_lptr;  // LDS int4*
#endif

union FragBF16 {
  v16bf v;
  uint4 u[2];   // u[0] = elements 0..7, u[1] = elements 8..15
};

__device__ __forceinline__ v8f vzero8() {
  v8f z;
#pragma unroll
  for (int i = 0; i < 8; ++i) z[i] = 0.0f;
  return z;
}

__device__ __forceinline__ v8f wmma_bf16(const v16bf& a, const v16bf& b, const v8f& c) {
  // (neg_a, A, neg_b, B, c_mod, C, reuse_a, reuse_b)
  return __builtin_amdgcn_wmma_f32_16x16x32_bf16(false, a, false, b, (short)0, c,
                                                 false, false);
}

// 16-byte global -> LDS copy: async (ASYNCcnt) when available, else via VGPRs.
__device__ __forceinline__ void copy16_g2l(__bf16* lds_dst, const __bf16* gsrc) {
#if HAVE_ASYNC_LDS
  __builtin_amdgcn_global_load_async_to_lds_b128((v4i_gptr)gsrc, (v4i_lptr)lds_dst,
                                                 0, 0);
#else
  *(uint4*)lds_dst = *(const uint4*)gsrc;
#endif
}

__device__ __forceinline__ void wait_async() {
#if HAVE_ASYNC_LDS
#if __has_builtin(__builtin_amdgcn_s_wait_asynccnt)
  __builtin_amdgcn_s_wait_asynccnt(0);
#else
  asm volatile("s_wait_asynccnt 0" ::: "memory");
#endif
#endif
}

// ---------------------------------------------------------------------------
// fp32 -> bf16 conversion (vectorized x4)
// ---------------------------------------------------------------------------
__global__ void cvt_f32_to_bf16(const float* __restrict__ s,
                                __bf16* __restrict__ d, int n4) {
  int i = blockIdx.x * blockDim.x + threadIdx.x;
  if (i >= n4) return;
  const float4 f = ((const float4*)s)[i];
  __bf16 o[4] = {(__bf16)f.x, (__bf16)f.y, (__bf16)f.z, (__bf16)f.w};
  *(ushort4*)(d + 4 * (size_t)i) = *(const ushort4*)o;
}

// ---------------------------------------------------------------------------
// RoPE, in place on bf16 [2048, H, 128]
// ---------------------------------------------------------------------------
__global__ void rope_inplace(__bf16* __restrict__ t,
                             const float* __restrict__ fc,
                             const float* __restrict__ fs, int H) {
  int idx = blockIdx.x * blockDim.x + threadIdx.x;
  int total = 2048 * H * 64;
  if (idx >= total) return;
  int i = idx & 63;
  int h = (idx >> 6) % H;
  int s = idx / (H * 64);
  size_t base = ((size_t)s * H + h) * 128 + 2 * i;
  float re = (float)t[base];
  float im = (float)t[base + 1];
  float c = fc[s * 64 + i];
  float sn = fs[s * 64 + i];
  t[base]     = (__bf16)(re * c - im * sn);
  t[base + 1] = (__bf16)(re * sn + im * c);
}

// ---------------------------------------------------------------------------
// NT GEMM: C[M,N] = A[M,K] * B[N,K]^T, A/B bf16 row-major (K contiguous).
// Block tile 128x256, BK=32, double-buffered async LDS staging.
// 8 waves: wave grid 2(M) x 4(N), each wave 64x64 = 4x4 WMMA tiles.
// ---------------------------------------------------------------------------
template <typename OUT_T>
__global__ __launch_bounds__(256) void gemm_nt_bf16(
    const __bf16* __restrict__ A, const __bf16* __restrict__ B,
    OUT_T* __restrict__ C, int M, int N, int K) {
  constexpr int LP = 40;                 // padded row stride (bf16 elems)
  constexpr int ASZ = 128 * LP;          // one A buffer
  constexpr int BSZ = 256 * LP;          // one B buffer
  __shared__ __bf16 sA[2 * ASZ];
  __shared__ __bf16 sB[2 * BSZ];

  const int tid  = threadIdx.x;
  const int wv   = tid >> 5;
  const int lane = tid & 31;
  const int ln   = lane & 15;
  const int half = lane >> 4;
  const int klo  = half * 8;             // K phase per documented A/B layout
  const int wm   = wv >> 2;              // 0..1
  const int wn   = wv & 3;               // 0..3
  const int bm   = blockIdx.y;
  const int bn   = blockIdx.x;

  v8f acc[4][4];
#pragma unroll
  for (int mt = 0; mt < 4; ++mt)
#pragma unroll
    for (int nt = 0; nt < 4; ++nt) acc[mt][nt] = vzero8();

  auto load_tiles = [&](int kk, int buf) {
    // A tile: 128 rows x 32 bf16 = 512 16B chunks; B tile: 256 rows = 1024.
#pragma unroll
    for (int c = 0; c < 2; ++c) {
      int chunk = tid + c * 256;
      int row = chunk >> 2;
      int col = (chunk & 3) * 8;
      copy16_g2l(sA + buf * ASZ + row * LP + col,
                 A + (size_t)(bm * 128 + row) * K + kk + col);
    }
#pragma unroll
    for (int c = 0; c < 4; ++c) {
      int chunk = tid + c * 256;
      int row = chunk >> 2;
      int col = (chunk & 3) * 8;
      copy16_g2l(sB + buf * BSZ + row * LP + col,
                 B + (size_t)(bn * 256 + row) * K + kk + col);
    }
  };

  const int nk = K >> 5;
  load_tiles(0, 0);
  wait_async();
  __syncthreads();

  for (int ki = 0; ki < nk; ++ki) {
    const int cur = ki & 1;
    if (ki + 1 < nk) load_tiles((ki + 1) << 5, cur ^ 1);

    const __bf16* bufA = sA + cur * ASZ;
    const __bf16* bufB = sB + cur * BSZ;
    FragBF16 af[4], bf[4];
#pragma unroll
    for (int mt = 0; mt < 4; ++mt) {
      const __bf16* p = bufA + (wm * 64 + mt * 16 + ln) * LP + klo;
      af[mt].u[0] = *(const uint4*)(p);
      af[mt].u[1] = *(const uint4*)(p + 16);
    }
#pragma unroll
    for (int nt = 0; nt < 4; ++nt) {
      const __bf16* p = bufB + (wn * 64 + nt * 16 + ln) * LP + klo;
      bf[nt].u[0] = *(const uint4*)(p);
      bf[nt].u[1] = *(const uint4*)(p + 16);
    }
#pragma unroll
    for (int mt = 0; mt < 4; ++mt)
#pragma unroll
      for (int nt = 0; nt < 4; ++nt)
        acc[mt][nt] = wmma_bf16(af[mt].v, bf[nt].v, acc[mt][nt]);

    wait_async();
    __syncthreads();
  }

  // C layout: vgpr r, lane l -> row = r + 8*half, col = l%16
#pragma unroll
  for (int mt = 0; mt < 4; ++mt) {
#pragma unroll
    for (int nt = 0; nt < 4; ++nt) {
#pragma unroll
      for (int r = 0; r < 8; ++r) {
        int row = bm * 128 + wm * 64 + mt * 16 + r + half * 8;
        int col = bn * 256 + wn * 64 + nt * 16 + ln;
        C[(size_t)row * N + col] = (OUT_T)acc[mt][nt][r];
      }
    }
  }
}

// ---------------------------------------------------------------------------
// Flash attention (causal, GQA 4:1). One block = (head h, 128-row Q tile).
// 8 waves, each owns 16 Q rows. K tiles of 64 keys streamed through LDS
// (async path); V stored transposed so the P@V B-operand is key-contiguous.
// ---------------------------------------------------------------------------
__global__ __launch_bounds__(256) void flash_attn(
    const __bf16* __restrict__ Q, const __bf16* __restrict__ K,
    const __bf16* __restrict__ V, __bf16* __restrict__ O) {
  constexpr int HD = 128, H = 32, KVH = 8;
  __shared__ __bf16 sK[64 * 136];        // [key][dim], padded
  __shared__ __bf16 sVt[128 * 72];       // [dim][key], padded
  __shared__ __bf16 sP[8][16 * 72];      // per-wave P staging [row][key]

  const int tid  = threadIdx.x;
  const int wv   = tid >> 5;
  const int lane = tid & 31;
  const int ln   = lane & 15;
  const int half = lane >> 4;
  const int klo  = half * 8;
  const int h    = blockIdx.x;           // 0..31
  const int qt   = blockIdx.y;           // 0..15
  const int g    = h >> 2;               // kv head

  // preload this wave's Q fragments (16 rows x 128 dims -> 4 A-frags)
  FragBF16 qf[4];
  const int qrow = qt * 128 + wv * 16 + ln;
  const __bf16* qp = Q + (size_t)qrow * (H * HD) + h * HD;
#pragma unroll
  for (int ks = 0; ks < 4; ++ks) {
    qf[ks].u[0] = *(const uint4*)(qp + ks * 32 + klo);
    qf[ks].u[1] = *(const uint4*)(qp + ks * 32 + klo + 16);
  }

  float m_r[8], l_r[8];
  v8f oacc[8];
#pragma unroll
  for (int r = 0; r < 8; ++r) { m_r[r] = -INFINITY; l_r[r] = 0.0f; }
#pragma unroll
  for (int dt = 0; dt < 8; ++dt) oacc[dt] = vzero8();

  const float scl = 0.08838834764831845f;  // 1/sqrt(128)
  const int nkt = 2 * qt + 2;              // causal bound in 64-key tiles

  for (int kt = 0; kt < nkt; ++kt) {
    // cooperative K/V tile load (64 keys x 128 dims each)
#pragma unroll
    for (int c = 0; c < 4; ++c) {
      int chunk = tid + c * 256;
      int krow  = chunk >> 4;              // 0..63
      int c8    = (chunk & 15) * 8;        // dim offset
      const size_t gbase = ((size_t)(kt * 64 + krow) * KVH + g) * HD + c8;
      copy16_g2l(sK + krow * 136 + c8, K + gbase);       // async path
      uint4 vvv = *(const uint4*)(V + gbase);            // sync + transpose
      const __bf16* ve = (const __bf16*)&vvv;
#pragma unroll
      for (int j = 0; j < 8; ++j) sVt[(c8 + j) * 72 + krow] = ve[j];
    }
    wait_async();
    __syncthreads();

    // S = (Q K^T) for this wave's 16 rows x 64 keys
    v8f sacc[4];
#pragma unroll
    for (int nt = 0; nt < 4; ++nt) {
      sacc[nt] = vzero8();
#pragma unroll
      for (int ks = 0; ks < 4; ++ks) {
        FragBF16 kf;
        const __bf16* p = sK + (nt * 16 + ln) * 136 + ks * 32 + klo;
        kf.u[0] = *(const uint4*)(p);
        kf.u[1] = *(const uint4*)(p + 16);
        sacc[nt] = wmma_bf16(qf[ks].v, kf.v, sacc[nt]);
      }
    }

    // scale + causal mask + per-row max (row lives across a 16-lane half)
    float cmax[8];
#pragma unroll
    for (int r = 0; r < 8; ++r) cmax[r] = -INFINITY;
#pragma unroll
    for (int nt = 0; nt < 4; ++nt) {
      const int kj = kt * 64 + nt * 16 + ln;
#pragma unroll
      for (int r = 0; r < 8; ++r) {
        const int qi = qt * 128 + wv * 16 + r + half * 8;
        float s = sacc[nt][r] * scl;
        s = (kj <= qi) ? s : -INFINITY;
        sacc[nt][r] = s;
        cmax[r] = fmaxf(cmax[r], s);
      }
    }
#pragma unroll
    for (int r = 0; r < 8; ++r)
#pragma unroll
      for (int off = 8; off >= 1; off >>= 1)
        cmax[r] = fmaxf(cmax[r], __shfl_xor(cmax[r], off, 32));

    float alpha[8], psum[8];
#pragma unroll
    for (int r = 0; r < 8; ++r) {
      const float mnew = fmaxf(m_r[r], cmax[r]);
      alpha[r] = __expf(m_r[r] - mnew);    // expf(-inf)=0 on first tile
      m_r[r]   = mnew;
      psum[r]  = 0.0f;
    }
    // exponentiate, accumulate row sums, stage P (bf16) in A-matrix layout
#pragma unroll
    for (int nt = 0; nt < 4; ++nt) {
#pragma unroll
      for (int r = 0; r < 8; ++r) {
        const float p = __expf(sacc[nt][r] - m_r[r]);
        psum[r] += p;
        sP[wv][(r + half * 8) * 72 + nt * 16 + ln] = (__bf16)p;
      }
    }
#pragma unroll
    for (int r = 0; r < 8; ++r) {
#pragma unroll
      for (int off = 8; off >= 1; off >>= 1)
        psum[r] += __shfl_xor(psum[r], off, 32);
      l_r[r] = l_r[r] * alpha[r] + psum[r];
    }
#pragma unroll
    for (int dt = 0; dt < 8; ++dt)
#pragma unroll
      for (int r = 0; r < 8; ++r) oacc[dt][r] = oacc[dt][r] * alpha[r];

    // O += P @ V  (A = P 16x64, B = V 64x128 via transposed LDS copy)
#pragma unroll
    for (int kps = 0; kps < 2; ++kps) {
      FragBF16 pf;
      const __bf16* pp = &sP[wv][ln * 72 + kps * 32 + klo];
      pf.u[0] = *(const uint4*)(pp);
      pf.u[1] = *(const uint4*)(pp + 16);
#pragma unroll
      for (int dt = 0; dt < 8; ++dt) {
        FragBF16 vf;
        const __bf16* vp = sVt + (dt * 16 + ln) * 72 + kps * 32 + klo;
        vf.u[0] = *(const uint4*)(vp);
        vf.u[1] = *(const uint4*)(vp + 16);
        oacc[dt] = wmma_bf16(pf.v, vf.v, oacc[dt]);
      }
    }
    __syncthreads();
  }

  // normalize and store ctx (bf16, [S, H*HD])
#pragma unroll
  for (int dt = 0; dt < 8; ++dt) {
#pragma unroll
    for (int r = 0; r < 8; ++r) {
      const int row = qt * 128 + wv * 16 + r + half * 8;
      const int col = h * HD + dt * 16 + ln;
      O[(size_t)row * (H * HD) + col] = (__bf16)(oacc[dt][r] / l_r[r]);
    }
  }
}

// ---------------------------------------------------------------------------
// Host launcher
// ---------------------------------------------------------------------------
extern "C" void kernel_launch(void* const* d_in, const int* in_sizes, int n_in,
                              void* d_out, int out_size, void* d_ws, size_t ws_size,
                              hipStream_t stream) {
  const float* x   = (const float*)d_in[0];
  const float* wq  = (const float*)d_in[1];
  const float* wk  = (const float*)d_in[2];
  const float* wvp = (const float*)d_in[3];
  const float* wo  = (const float*)d_in[4];
  const float* fc  = (const float*)d_in[5];
  const float* fs  = (const float*)d_in[6];
  float* out = (float*)d_out;

  char* ws = (char*)d_ws;
  size_t off = 0;
  auto alloc = [&](size_t bytes) -> char* {
    char* p = ws + off;
    off += (bytes + 255) & ~(size_t)255;
    return p;
  };
  __bf16* xb  = (__bf16*)alloc(2048ull * 4096 * 2);
  __bf16* wqb = (__bf16*)alloc(4096ull * 4096 * 2);
  __bf16* wkb = (__bf16*)alloc(1024ull * 4096 * 2);
  __bf16* wvb = (__bf16*)alloc(1024ull * 4096 * 2);
  __bf16* wob = (__bf16*)alloc(4096ull * 4096 * 2);
  __bf16* qb  = (__bf16*)alloc(2048ull * 4096 * 2);
  __bf16* kb  = (__bf16*)alloc(2048ull * 1024 * 2);
  __bf16* vb  = (__bf16*)alloc(2048ull * 1024 * 2);
  __bf16* cb  = (__bf16*)alloc(2048ull * 4096 * 2);

  auto cvt = [&](const float* s, __bf16* d, int n) {
    int n4 = n >> 2;
    cvt_f32_to_bf16<<<(n4 + 255) / 256, 256, 0, stream>>>(s, d, n4);
  };
  cvt(x,   xb,  2048 * 4096);
  cvt(wq,  wqb, 4096 * 4096);
  cvt(wk,  wkb, 1024 * 4096);
  cvt(wvp, wvb, 1024 * 4096);
  cvt(wo,  wob, 4096 * 4096);

  // Projections: C[M,N] = A[M,K] * B[N,K]^T   (block tile 128x256)
  gemm_nt_bf16<__bf16><<<dim3(16, 16), 256, 0, stream>>>(xb, wqb, qb, 2048, 4096, 4096);
  gemm_nt_bf16<__bf16><<<dim3(4, 16),  256, 0, stream>>>(xb, wkb, kb, 2048, 1024, 4096);
  gemm_nt_bf16<__bf16><<<dim3(4, 16),  256, 0, stream>>>(xb, wvb, vb, 2048, 1024, 4096);

  rope_inplace<<<(2048 * 32 * 64 + 255) / 256, 256, 0, stream>>>(qb, fc, fs, 32);
  rope_inplace<<<(2048 * 8 * 64 + 255) / 256, 256, 0, stream>>>(kb, fc, fs, 8);

  flash_attn<<<dim3(32, 16), 256, 0, stream>>>(qb, kb, vb, cb);

  // out = ctx @ wo^T (fp32 output)
  gemm_nt_bf16<float><<<dim3(16, 16), 256, 0, stream>>>(cb, wob, out, 2048, 4096, 4096);
}
